// GaussionConvolution_F_49838800503664
// MI455X (gfx1250) — matrix operand order repaired and checked by reading, not compiled, not run on hardware
//
#include <hip/hip_runtime.h>
#include <hip/hip_bf16.h>
#include <stdint.h>

#define N_NODES 100000
#define F_DIM   256
#define UNITS   256
#define DIMH    128
#define GAMMA   1.0f
#define LDK     264   // padded LDS row stride (bf16 elems) for K=256; 528B, 8B-aligned

typedef __attribute__((ext_vector_type(16))) __bf16 v16bf;
typedef __attribute__((ext_vector_type(8)))  float  v8f;

__device__ __forceinline__ unsigned short f32_to_bf16(float f) {
    union { float f; uint32_t u; } x; x.f = f;
    uint32_t u = x.u + 0x7FFFu + ((x.u >> 16) & 1u);   // round-to-nearest-even
    return (unsigned short)(u >> 16);
}
__device__ __forceinline__ float bflo(uint32_t u) {
    union { uint32_t u; float f; } x; x.u = u << 16; return x.f;
}
__device__ __forceinline__ float bfhi(uint32_t u) {
    union { uint32_t u; float f; } x; x.u = u & 0xFFFF0000u; return x.f;
}

// ---------------------------------------------------------------------------
// Fused GEMM (bf16 WMMA, f32 accum) + elu/relu/att epilogue + KL reduction.
// Stage W rows k=0..63 (64x256 bf16, transposed) + the block's 32-row A slab into
// LDS once (152 KB of the 320 KB WGP LDS -> 2 blocks/WGP), one barrier, then
// a fully unrolled K loop of pure ds_load + v_wmma.
// Grid: N/32 blocks of 256 threads (8 waves). Wave = (rowtile rt, colgroup cg).
// Wave accumulators: mean cols [32cg,32cg+32) and var cols [128+32cg, ...+32).
// ---------------------------------------------------------------------------
__global__ __launch_bounds__(256)
void gemm_epilogue_kernel(const float* __restrict__ A, const float* __restrict__ W,
                          unsigned short* __restrict__ p1, unsigned short* __restrict__ p2,
                          float* __restrict__ kl_out) {
    __shared__ unsigned short ldsA[32 * LDK];     // [row][k]  bf16, full K
    __shared__ unsigned short ldsB[256 * LDK];    // [n][k]    bf16 (W transposed), full K

    const int tid  = threadIdx.x;
    const int lane = tid & 31;
    const int wave = tid >> 5;
    const int rt   = wave & 1;          // which 16-row tile of the 32
    const int cg   = wave >> 1;         // 0..3 -> 32-column group
    const int row0 = blockIdx.x * 32;
    const int hi   = lane >> 4;         // half-lane group
    const int m16  = lane & 15;

    // ---- stage A slab: 32 rows x 256 k, f32 -> bf16 (2048 float4) ----
    #pragma unroll
    for (int rep = 0; rep < 8; ++rep) {
        int linear = rep * 256 + tid;
        int r  = linear >> 6;                 // 0..31
        int kq = (linear & 63) << 2;          // 0..252
        float4 a4 = *(const float4*)(A + (size_t)(row0 + r) * F_DIM + kq);
        unsigned short* dst = &ldsA[r * LDK + kq];
        dst[0] = f32_to_bf16(a4.x); dst[1] = f32_to_bf16(a4.y);
        dst[2] = f32_to_bf16(a4.z); dst[3] = f32_to_bf16(a4.w);
    }
    // ---- stage W rows k=0..63 transposed: ldsB[n][k] (4096 float4) ----
    #pragma unroll
    for (int rep = 0; rep < 16; ++rep) {
        int linear = rep * 256 + tid;
        int k  = linear >> 6;                 // 0..63
        int nq = (linear & 63) << 2;          // 0..252
        float4 w4 = *(const float4*)(W + (size_t)k * UNITS + nq);
        ldsB[(nq + 0) * LDK + k] = f32_to_bf16(w4.x);
        ldsB[(nq + 1) * LDK + k] = f32_to_bf16(w4.y);
        ldsB[(nq + 2) * LDK + k] = f32_to_bf16(w4.z);
        ldsB[(nq + 3) * LDK + k] = f32_to_bf16(w4.w);
    }
    __syncthreads();

    v8f accm0 = {}, accm1 = {}, accv0 = {}, accv1 = {};
    const int arow = (rt * 16 + m16) * LDK;
    const int nm   = cg * 32 + m16;

    #pragma unroll
    for (int k0 = 0; k0 < F_DIM; k0 += 32) {
        union { v16bf v; uint32_t u[8]; } af, bm0, bm1, bv0, bv1;
        #pragma unroll
        for (int v = 0; v < 8; ++v) {
            // A 16x32 bf16 striping: lanes 0-15 K=0..15, lanes 16-31 K=16..31
            int kbA = k0 + (hi << 4) + (v << 1);
            // B 32x16 striping: lanes 0-15 K=0..15, lanes 16-31 K=16..31
            int kbB = k0 + (hi << 4) + (v << 1);
            af.u[v]  = *(const uint32_t*)&ldsA[arow + kbA];
            bm0.u[v] = *(const uint32_t*)&ldsB[(nm           ) * LDK + kbB];
            bm1.u[v] = *(const uint32_t*)&ldsB[(nm + 16      ) * LDK + kbB];
            bv0.u[v] = *(const uint32_t*)&ldsB[(nm + DIMH    ) * LDK + kbB];
            bv1.u[v] = *(const uint32_t*)&ldsB[(nm + DIMH+16 ) * LDK + kbB];
        }
        accm0 = __builtin_amdgcn_wmma_f32_16x16x32_bf16(false, af.v, false, bm0.v, (short)0, accm0, false, false);
        accm1 = __builtin_amdgcn_wmma_f32_16x16x32_bf16(false, af.v, false, bm1.v, (short)0, accm1, false, false);
        accv0 = __builtin_amdgcn_wmma_f32_16x16x32_bf16(false, af.v, false, bv0.v, (short)0, accv0, false, false);
        accv1 = __builtin_amdgcn_wmma_f32_16x16x32_bf16(false, af.v, false, bv1.v, (short)0, accv1, false, false);
    }

    // ---- fused epilogue: elu/relu, attention, KL, bf16 staging stores ----
    float klacc = 0.0f;
    const int nodebase = row0 + rt * 16;
    #pragma unroll
    for (int j = 0; j < 8; ++j) {
        const int node = nodebase + (hi ? j + 8 : j);   // C/D layout: M = j (+8 for hi lanes)
        #pragma unroll
        for (int t = 0; t < 2; ++t) {
            float hm = t ? accm1[j] : accm0[j];
            float hv = t ? accv1[j] : accv0[j];
            float mean = hm > 0.0f ? hm : (__expf(hm) - 1.0f);   // elu
            float var  = hv > 0.0f ? hv : 0.0f;                  // relu
            float att  = __expf(-GAMMA * var);
            int d = cg * 32 + t * 16 + m16;
            p1[(size_t)node * DIMH + d] = f32_to_bf16(mean * att);
            p2[(size_t)node * DIMH + d] = f32_to_bf16(var * att * att);
            klacc += mean * mean + var - __logf(1e-8f + var) - 1.0f;
        }
    }
    klacc *= 0.5f / (float)DIMH;
    #pragma unroll
    for (int off = 16; off > 0; off >>= 1)
        klacc += __shfl_xor(klacc, off, 32);
    if (lane == 0) atomicAdd(kl_out, klacc);
}

// ---------------------------------------------------------------------------
// CSR row pointers from sorted row[] via per-node lower_bound binary search.
// ---------------------------------------------------------------------------
__global__ void rowptr_kernel(const int* __restrict__ row, int* __restrict__ rp,
                              int n, int e) {
    int i = blockIdx.x * blockDim.x + threadIdx.x;
    if (i > n) return;
    if (i == n) { rp[n] = e; return; }
    int lo = 0, hh = e;
    while (lo < hh) { int mid = (lo + hh) >> 1; if (row[mid] < i) lo = mid + 1; else hh = mid; }
    rp[i] = lo;
}

// ---------------------------------------------------------------------------
// Dual SpMM segment reduction: one wave per node, lane owns 4 dims (float4).
// bf16 gathers are L2-resident (2 x 25.6 MB tables); f32 accumulation.
// 4-edge manual unroll -> 8 independent 8B gathers in flight per iteration.
// ---------------------------------------------------------------------------
__global__ __launch_bounds__(256)
void spmm_kernel(const int* __restrict__ rp, const int* __restrict__ col,
                 const float* __restrict__ a1, const float* __restrict__ a2,
                 const unsigned short* __restrict__ p1, const unsigned short* __restrict__ p2,
                 float* __restrict__ out) {
    const int lane = threadIdx.x & 31;
    const int node = (blockIdx.x * blockDim.x + threadIdx.x) >> 5;
    if (node >= N_NODES) return;
    const int e0 = rp[node], e1 = rp[node + 1];
    float m0 = 0, m1 = 0, m2 = 0, m3 = 0;
    float v0 = 0, v1 = 0, v2 = 0, v3 = 0;
    const int loff = lane * 4;

    int e = e0;
    for (; e + 4 <= e1; e += 4) {
        int   c0 = col[e],  c1 = col[e + 1], c2 = col[e + 2], c3 = col[e + 3];
        float wa0 = a1[e], wa1 = a1[e + 1], wa2 = a1[e + 2], wa3 = a1[e + 3];
        float wb0 = a2[e], wb1 = a2[e + 1], wb2 = a2[e + 2], wb3 = a2[e + 3];
        uint2 qa0 = *(const uint2*)(p1 + (size_t)c0 * DIMH + loff);
        uint2 qa1 = *(const uint2*)(p1 + (size_t)c1 * DIMH + loff);
        uint2 qa2 = *(const uint2*)(p1 + (size_t)c2 * DIMH + loff);
        uint2 qa3 = *(const uint2*)(p1 + (size_t)c3 * DIMH + loff);
        uint2 qb0 = *(const uint2*)(p2 + (size_t)c0 * DIMH + loff);
        uint2 qb1 = *(const uint2*)(p2 + (size_t)c1 * DIMH + loff);
        uint2 qb2 = *(const uint2*)(p2 + (size_t)c2 * DIMH + loff);
        uint2 qb3 = *(const uint2*)(p2 + (size_t)c3 * DIMH + loff);
        m0 += wa0 * bflo(qa0.x); m1 += wa0 * bfhi(qa0.x); m2 += wa0 * bflo(qa0.y); m3 += wa0 * bfhi(qa0.y);
        m0 += wa1 * bflo(qa1.x); m1 += wa1 * bfhi(qa1.x); m2 += wa1 * bflo(qa1.y); m3 += wa1 * bfhi(qa1.y);
        m0 += wa2 * bflo(qa2.x); m1 += wa2 * bfhi(qa2.x); m2 += wa2 * bflo(qa2.y); m3 += wa2 * bfhi(qa2.y);
        m0 += wa3 * bflo(qa3.x); m1 += wa3 * bfhi(qa3.x); m2 += wa3 * bflo(qa3.y); m3 += wa3 * bfhi(qa3.y);
        v0 += wb0 * bflo(qb0.x); v1 += wb0 * bfhi(qb0.x); v2 += wb0 * bflo(qb0.y); v3 += wb0 * bfhi(qb0.y);
        v0 += wb1 * bflo(qb1.x); v1 += wb1 * bfhi(qb1.x); v2 += wb1 * bflo(qb1.y); v3 += wb1 * bfhi(qb1.y);
        v0 += wb2 * bflo(qb2.x); v1 += wb2 * bfhi(qb2.x); v2 += wb2 * bflo(qb2.y); v3 += wb2 * bfhi(qb2.y);
        v0 += wb3 * bflo(qb3.x); v1 += wb3 * bfhi(qb3.x); v2 += wb3 * bflo(qb3.y); v3 += wb3 * bfhi(qb3.y);
    }
    for (; e < e1; ++e) {
        const int   c  = col[e];
        const float w1 = a1[e];
        const float w2 = a2[e];
        const uint2 q1 = *(const uint2*)(p1 + (size_t)c * DIMH + loff);
        const uint2 q2 = *(const uint2*)(p2 + (size_t)c * DIMH + loff);
        m0 += w1 * bflo(q1.x); m1 += w1 * bfhi(q1.x);
        m2 += w1 * bflo(q1.y); m3 += w1 * bfhi(q1.y);
        v0 += w2 * bflo(q2.x); v1 += w2 * bfhi(q2.x);
        v2 += w2 * bflo(q2.y); v3 += w2 * bfhi(q2.y);
    }
    float4 om = { m0, m1, m2, m3 };
    float4 ov = { v0, v1, v2, v3 };
    *(float4*)(out + (size_t)node * UNITS + loff)        = om;
    *(float4*)(out + (size_t)node * UNITS + DIMH + loff) = ov;
}

__global__ void zero_kl_kernel(float* kl) { *kl = 0.0f; }

extern "C" void kernel_launch(void* const* d_in, const int* in_sizes, int n_in,
                              void* d_out, int out_size, void* d_ws, size_t ws_size,
                              hipStream_t stream) {
    const float* feats = (const float*)d_in[0];
    const float* W     = (const float*)d_in[1];
    const int*   row   = (const int*)d_in[2];
    const int*   col   = (const int*)d_in[3];
    const float* a1    = (const float*)d_in[4];
    const float* a2    = (const float*)d_in[5];
    float*       out   = (float*)d_out;
    const int    E     = in_sizes[2];

    // workspace: p1 (N*128 bf16) | p2 (N*128 bf16) | row_ptr (N+1 ints)
    unsigned short* p1 = (unsigned short*)d_ws;
    unsigned short* p2 = p1 + (size_t)N_NODES * DIMH;
    int*            rp = (int*)(p2 + (size_t)N_NODES * DIMH);
    float*         klp = out + (size_t)N_NODES * UNITS;   // kl scalar after out matrix

    zero_kl_kernel<<<1, 1, 0, stream>>>(klp);
    gemm_epilogue_kernel<<<N_NODES / 32, 256, 0, stream>>>(feats, W, p1, p2, klp);
    rowptr_kernel<<<(N_NODES + 256) / 256, 256, 0, stream>>>(row, rp, N_NODES, E);
    spmm_kernel<<<(N_NODES * 32 + 255) / 256, 256, 0, stream>>>(rp, col, a1, a2, p1, p2, out);
}
